// ProhibitusModel_79508434583852
// MI455X (gfx1250) — matrix operand order repaired and checked by reading, not compiled, not run on hardware
//
#include <hip/hip_runtime.h>
#include <hip/hip_bf16.h>
#include <math.h>

// Problem dims (fixed by the reference)
#define BQ   2
#define TT   2048
#define DD   512
#define HH   8
#define LL   6
#define VV   32000
#define HDIM 64
#define MM   (BQ * TT)   // 4096 rows
#define QT_PER_SEQ (TT / 16)   // 128 query tiles per sequence

typedef __attribute__((ext_vector_type(16))) __bf16 v16bf;
typedef __attribute__((ext_vector_type(8)))  __bf16 v8bf;
typedef __attribute__((ext_vector_type(4)))  __bf16 v4bf;
typedef __attribute__((ext_vector_type(8)))  float  v8f;

enum { ACT_NONE = 0, ACT_GELU = 1 };
enum { OUT_F32 = 0, OUT_BF16 = 1, OUT_BF16_T = 2 };

#define SHUF16(lo, hi) __builtin_shufflevector(lo, hi, \
        0,1,2,3,4,5,6,7,8,9,10,11,12,13,14,15)

__device__ __forceinline__ float wave_allreduce_sum(float v) {
#pragma unroll
    for (int off = 16; off > 0; off >>= 1)
        v += __shfl_xor(v, off, 32);
    return v;
}

// ---------------------------------------------------------------------------
// fp32 -> bf16 bulk conversion (vectorized x4)
// ---------------------------------------------------------------------------
__global__ void cvt_bf16_kernel(const float* __restrict__ in,
                                __bf16* __restrict__ out, long n4) {
    long i = (long)blockIdx.x * blockDim.x + threadIdx.x;
    if (i >= n4) return;
    float4 f = ((const float4*)in)[i];
    v4bf o;
    o[0] = (__bf16)f.x; o[1] = (__bf16)f.y;
    o[2] = (__bf16)f.z; o[3] = (__bf16)f.w;
    ((v4bf*)out)[i] = o;
}

// ---------------------------------------------------------------------------
// Embedding: x[m, d] = tok_emb[idx[m], d] + pos_emb[m % T, d]   (fp32 out)
// ---------------------------------------------------------------------------
__global__ void embed_kernel(const int* __restrict__ idx,
                             const float* __restrict__ tok,
                             const float* __restrict__ pos,
                             float* __restrict__ x) {
    int i = blockIdx.x * blockDim.x + threadIdx.x;
    if (i >= MM * DD) return;
    int m = i / DD;
    int d = i - m * DD;
    int t = m % TT;
    x[i] = tok[(size_t)idx[m] * DD + d] + pos[(size_t)t * DD + d];
}

// ---------------------------------------------------------------------------
// LayerNorm: one wave32 per row of D=512; bf16 output (feeds WMMA GEMMs)
// ---------------------------------------------------------------------------
__global__ __launch_bounds__(256) void ln_kernel(const float* __restrict__ x,
                                                 const float* __restrict__ w,
                                                 const float* __restrict__ b,
                                                 __bf16* __restrict__ y) {
    int lane = threadIdx.x;            // 0..31
    int row  = blockIdx.x * blockDim.y + threadIdx.y;
    if (row >= MM) return;
    const float* xr = x + (size_t)row * DD;
    float vals[DD / 32];
    float s = 0.f, s2 = 0.f;
#pragma unroll
    for (int j = 0; j < DD / 32; ++j) {
        float v = xr[lane + j * 32];
        vals[j] = v;
        s += v;
        s2 += v * v;
    }
    s  = wave_allreduce_sum(s);
    s2 = wave_allreduce_sum(s2);
    const float inv_d = 1.0f / (float)DD;
    float mean = s * inv_d;
    float var  = s2 * inv_d - mean * mean;
    float rstd = rsqrtf(var + 1e-5f);
    __bf16* yr = y + (size_t)row * DD;
#pragma unroll
    for (int j = 0; j < DD / 32; ++j) {
        int d = lane + j * 32;
        yr[d] = (__bf16)((vals[j] - mean) * rstd * w[d] + b[d]);
    }
}

// ---------------------------------------------------------------------------
// WMMA GEMM, bf16 inputs, f32 accumulate:
//   Y[M,N] = act(Xbf[M,K] @ Wbf[N,K]^T + bias) (+ resid)
// 32x64 tile per wave32 -> 8 WMMAs per K-step of 32.
// out_mode: OUT_F32 / OUT_BF16 (row major) / OUT_BF16_T (stores Y^T [N,M])
// ---------------------------------------------------------------------------
#define GEMM_WAVES 8

__global__ __launch_bounds__(256) void wmma_gemm_bf16_kernel(
        const __bf16* __restrict__ X, const __bf16* __restrict__ W,
        const float* __restrict__ bias, const float* __restrict__ resid,
        void* __restrict__ Yv, int Mdim, int Ndim, int Kdim,
        int act, int out_mode) {
    int lane   = threadIdx.x & 31;
    int wave   = threadIdx.y;
    int tilesN = Ndim >> 6;                      // 64-wide N tiles
    int tileId = blockIdx.x * GEMM_WAVES + wave;
    int tileM  = tileId / tilesN;
    int tileN  = tileId - tileM * tilesN;
    if (tileM >= (Mdim >> 5)) return;            // 32-tall M tiles

    int lane16 = lane & 15;
    int kh     = lane >> 4;

    const __bf16* A0 = X + (size_t)(tileM * 32 + lane16) * Kdim;
    const __bf16* A1 = A0 + (size_t)16 * Kdim;
    const __bf16* B0 = W + (size_t)(tileN * 64 + lane16) * Kdim;
    const size_t  Bstep = (size_t)16 * Kdim;     // 16 cols of W

    v8f c[2][4];
#pragma unroll
    for (int mi = 0; mi < 2; ++mi)
#pragma unroll
        for (int ni = 0; ni < 4; ++ni)
            c[mi][ni] = (v8f){};

    for (int kk = 0; kk < Kdim; kk += 32) {
        int abase = kk + kh * 16;
        v8bf a0lo = *(const v8bf*)(A0 + abase);
        v8bf a0hi = *(const v8bf*)(A0 + abase + 8);
        v8bf a1lo = *(const v8bf*)(A1 + abase);
        v8bf a1hi = *(const v8bf*)(A1 + abase + 8);
        v16bf amat[2];
        amat[0] = SHUF16(a0lo, a0hi);
        amat[1] = SHUF16(a1lo, a1hi);

        int bbase = kk + kh * 16;
        v16bf bmat[4];
#pragma unroll
        for (int ni = 0; ni < 4; ++ni)
            bmat[ni] = *(const v16bf*)(B0 + (size_t)ni * Bstep + bbase);

#pragma unroll
        for (int mi = 0; mi < 2; ++mi)
#pragma unroll
            for (int ni = 0; ni < 4; ++ni)
                c[mi][ni] = __builtin_amdgcn_wmma_f32_16x16x32_bf16(
                        false, amat[mi], false, bmat[ni],
                        (short)0, c[mi][ni], false, false);
    }

    // ---- epilogue: bias, activation, residual, store ----
#pragma unroll
    for (int ni = 0; ni < 4; ++ni) {
        int n  = tileN * 64 + ni * 16 + lane16;
        float bn = bias ? bias[n] : 0.0f;
#pragma unroll
        for (int mi = 0; mi < 2; ++mi) {
            int mbase = tileM * 32 + mi * 16 + 8 * kh;
#pragma unroll
            for (int r = 0; r < 8; ++r) {
                int m = mbase + r;
                float v = c[mi][ni][r] + bn;
                if (act == ACT_GELU)
                    v = 0.5f * v * (1.0f + erff(v * 0.70710678118654752f));
                if (resid) v += resid[(size_t)m * Ndim + n];
                if (out_mode == OUT_BF16_T)
                    ((__bf16*)Yv)[(size_t)n * Mdim + m] = (__bf16)v;
                else if (out_mode == OUT_BF16)
                    ((__bf16*)Yv)[(size_t)m * Ndim + n] = (__bf16)v;
                else
                    ((float*)Yv)[(size_t)m * Ndim + n] = v;
            }
        }
    }
}

// ---------------------------------------------------------------------------
// WMMA flash attention (causal, online softmax).
// One wave32 per (b, h, 16-query tile). Keys processed in chunks of 32:
//   S(16x32) via 4 WMMAs (Q:16x64 @ K^T), softmax in f32 with per-row
//   stats held per half-wave (C/D row layout), P restaged through LDS to
//   the A-fragment layout, then O(16x64) += P @ V via 4 WMMAs using the
//   transposed V buffer [D, M] for contiguous B fragments.
// ---------------------------------------------------------------------------
#define AT_WAVES 8

__global__ __launch_bounds__(256) void flash_attn_kernel(
        const __bf16* __restrict__ q,   // [M, D]
        const __bf16* __restrict__ k,   // [M, D]
        const __bf16* __restrict__ vt,  // [D, M]  (V transposed)
        __bf16* __restrict__ y) {       // [M, D]
    __shared__ __bf16 pbuf[AT_WAVES][16][32];

    int lane = threadIdx.x & 31;
    int wave = threadIdx.y;
    int qid  = blockIdx.x * AT_WAVES + wave;    // 0 .. B*H*(T/16)-1
    int qt   = qid & (QT_PER_SEQ - 1);
    int bh   = qid >> 7;                        // QT_PER_SEQ == 128
    int h    = bh & (HH - 1);
    int b    = bh >> 3;

    int lane16 = lane & 15;
    int kh     = lane >> 4;

    // Q fragments (two 16x32 A fragments covering HD=64)
    const __bf16* qrow = q + (size_t)(b * TT + qt * 16 + lane16) * DD + h * HDIM;
    v16bf aq[2];
#pragma unroll
    for (int s = 0; s < 2; ++s) {
        v8bf lo = *(const v8bf*)(qrow + s * 32 + kh * 16);
        v8bf hi = *(const v8bf*)(qrow + s * 32 + kh * 16 + 8);
        aq[s] = SHUF16(lo, hi);
    }

    v8f o[4];
#pragma unroll
    for (int nd = 0; nd < 4; ++nd) o[nd] = (v8f){};
    float mrow[8], lrow[8];
#pragma unroll
    for (int r = 0; r < 8; ++r) { mrow[r] = -INFINITY; lrow[r] = 0.0f; }

    const float scale = 0.125f;            // 1/sqrt(64)
    const int   tqbase = qt * 16 + 8 * kh; // +r gives this lane-half's query pos
    const int   nch = (qt * 16 + 47) >> 5; // ceil((qt*16+16)/32)
    const __bf16* kbase = k + (size_t)(b * TT) * DD + h * HDIM;

    for (int c = 0; c < nch; ++c) {
        int j0 = c * 32;
        // ---- S = scale * Q @ K^T for 32 keys (two 16-key column tiles) ----
        v8f s0 = (v8f){}, s1 = (v8f){};
#pragma unroll
        for (int s = 0; s < 2; ++s) {
            v16bf bk0 = *(const v16bf*)(kbase + (size_t)(j0 + lane16) * DD
                                        + s * 32 + kh * 16);
            v16bf bk1 = *(const v16bf*)(kbase + (size_t)(j0 + 16 + lane16) * DD
                                        + s * 32 + kh * 16);
            s0 = __builtin_amdgcn_wmma_f32_16x16x32_bf16(false, aq[s], false, bk0,
                                                         (short)0, s0, false, false);
            s1 = __builtin_amdgcn_wmma_f32_16x16x32_bf16(false, aq[s], false, bk1,
                                                         (short)0, s1, false, false);
        }

        bool need_mask = (j0 + 31) > (qt * 16);   // wave-uniform

        // ---- online softmax, per output row (8 rows per half-wave) ----
#pragma unroll
        for (int r = 0; r < 8; ++r) {
            float e0 = s0[r] * scale;
            float e1 = s1[r] * scale;
            if (need_mask) {
                int tq = tqbase + r;
                if (j0 + lane16 > tq)      e0 = -INFINITY;
                if (j0 + 16 + lane16 > tq) e1 = -INFINITY;
            }
            float rmax = fmaxf(e0, e1);
#pragma unroll
            for (int off = 8; off > 0; off >>= 1)
                rmax = fmaxf(rmax, __shfl_xor(rmax, off, 32));
            float mnew = fmaxf(mrow[r], rmax);
            float p0 = __expf(e0 - mnew);
            float p1 = __expf(e1 - mnew);
            float rs = p0 + p1;
#pragma unroll
            for (int off = 8; off > 0; off >>= 1)
                rs += __shfl_xor(rs, off, 32);
            float corr = __expf(mrow[r] - mnew);  // 0 on first chunk
            lrow[r] = lrow[r] * corr + rs;
            mrow[r] = mnew;
#pragma unroll
            for (int nd = 0; nd < 4; ++nd) o[nd][r] *= corr;
            int m = r + 8 * kh;
            pbuf[wave][m][lane16]      = (__bf16)p0;
            pbuf[wave][m][lane16 + 16] = (__bf16)p1;
        }

        // ---- restage P into A-fragment layout via LDS ----
        v8bf plo = *(const v8bf*)&pbuf[wave][lane16][kh * 16];
        v8bf phi = *(const v8bf*)&pbuf[wave][lane16][kh * 16 + 8];
        v16bf pfrag = SHUF16(plo, phi);

        // ---- O += P @ V (V^T rows give contiguous B fragments) ----
#pragma unroll
        for (int nd = 0; nd < 4; ++nd) {
            const __bf16* vrow = vt + (size_t)(h * HDIM + nd * 16 + lane16) * MM
                                 + b * TT + j0 + kh * 16;
            v16bf bv = *(const v16bf*)vrow;
            o[nd] = __builtin_amdgcn_wmma_f32_16x16x32_bf16(false, pfrag, false, bv,
                                                            (short)0, o[nd], false, false);
        }
    }

    // ---- normalize and store ----
    float invl[8];
#pragma unroll
    for (int r = 0; r < 8; ++r) invl[r] = 1.0f / lrow[r];
    __bf16* ybase = y + (size_t)(b * TT + qt * 16) * DD + h * HDIM;
#pragma unroll
    for (int nd = 0; nd < 4; ++nd) {
        int hd = nd * 16 + lane16;
#pragma unroll
        for (int r = 0; r < 8; ++r) {
            int m = r + 8 * kh;
            ybase[(size_t)m * DD + hd] = (__bf16)(o[nd][r] * invl[r]);
        }
    }
}

// ---------------------------------------------------------------------------
// Host-side orchestration
// ---------------------------------------------------------------------------
static inline void launch_gemm(const __bf16* X, const __bf16* W,
                               const float* bias, const float* resid, void* Y,
                               int M, int N, int K, int act, int out_mode,
                               hipStream_t stream) {
    int tiles  = (M / 32) * (N / 64);
    int blocks = (tiles + GEMM_WAVES - 1) / GEMM_WAVES;
    wmma_gemm_bf16_kernel<<<blocks, dim3(32, GEMM_WAVES), 0, stream>>>(
        X, W, bias, resid, Y, M, N, K, act, out_mode);
}

static inline void launch_cvt(const float* in, __bf16* out, long n,
                              hipStream_t stream) {
    long n4 = n / 4;
    int blocks = (int)((n4 + 255) / 256);
    cvt_bf16_kernel<<<blocks, 256, 0, stream>>>(in, out, n4);
}

extern "C" void kernel_launch(void* const* d_in, const int* in_sizes, int n_in,
                              void* d_out, int out_size, void* d_ws, size_t ws_size,
                              hipStream_t stream) {
    const int*   idx     = (const int*)  d_in[0];
    const float* tok_emb = (const float*)d_in[1];
    const float* pos_emb = (const float*)d_in[2];
    const float* Wq      = (const float*)d_in[3];
    const float* bq      = (const float*)d_in[4];
    const float* Wk      = (const float*)d_in[5];
    const float* bk      = (const float*)d_in[6];
    const float* Wv      = (const float*)d_in[7];
    const float* bv      = (const float*)d_in[8];
    const float* Wo      = (const float*)d_in[9];
    const float* bo      = (const float*)d_in[10];
    const float* ln1_w   = (const float*)d_in[11];
    const float* ln1_b   = (const float*)d_in[12];
    const float* ln2_w   = (const float*)d_in[13];
    const float* ln2_b   = (const float*)d_in[14];
    const float* W1      = (const float*)d_in[15];
    const float* b1      = (const float*)d_in[16];
    const float* W2      = (const float*)d_in[17];
    const float* b2      = (const float*)d_in[18];
    const float* ln_f_w  = (const float*)d_in[19];
    const float* ln_f_b  = (const float*)d_in[20];
    const float* head_W  = (const float*)d_in[21];

    // ---- workspace layout ----
    const size_t MD    = (size_t)MM * DD;           // 4096*512
    const size_t WSQ   = (size_t)LL * DD * DD;
    const size_t WMLP  = (size_t)LL * 4 * DD * DD;
    const size_t WHEAD = (size_t)VV * DD;

    char* p = (char*)d_ws;
    __bf16* wq_bf = (__bf16*)p;      p += WSQ  * sizeof(__bf16);
    __bf16* wk_bf = (__bf16*)p;      p += WSQ  * sizeof(__bf16);
    __bf16* wv_bf = (__bf16*)p;      p += WSQ  * sizeof(__bf16);
    __bf16* wo_bf = (__bf16*)p;      p += WSQ  * sizeof(__bf16);
    __bf16* w1_bf = (__bf16*)p;      p += WMLP * sizeof(__bf16);
    __bf16* w2_bf = (__bf16*)p;      p += WMLP * sizeof(__bf16);
    __bf16* wh_bf = (__bf16*)p;      p += WHEAD * sizeof(__bf16);
    float*  x     = (float*)p;       p += MD * sizeof(float);
    __bf16* xn    = (__bf16*)p;      p += MD * sizeof(__bf16);
    __bf16* qbf   = (__bf16*)p;      p += MD * sizeof(__bf16);
    __bf16* kbf   = (__bf16*)p;      p += MD * sizeof(__bf16);
    __bf16* vbt   = (__bf16*)p;      p += MD * sizeof(__bf16);  // [D, M]
    __bf16* yb    = (__bf16*)p;      p += MD * sizeof(__bf16);
    __bf16* hb    = (__bf16*)p;      /* [M, 4D] */

    // ---- weight down-conversion to bf16 (L2-resident working set) ----
    launch_cvt(Wq, wq_bf, (long)WSQ,  stream);
    launch_cvt(Wk, wk_bf, (long)WSQ,  stream);
    launch_cvt(Wv, wv_bf, (long)WSQ,  stream);
    launch_cvt(Wo, wo_bf, (long)WSQ,  stream);
    launch_cvt(W1, w1_bf, (long)WMLP, stream);
    launch_cvt(W2, w2_bf, (long)WMLP, stream);
    launch_cvt(head_W, wh_bf, (long)WHEAD, stream);

    // ---- embedding ----
    embed_kernel<<<(MM * DD + 255) / 256, 256, 0, stream>>>(idx, tok_emb, pos_emb, x);

    const size_t DSq = (size_t)DD * DD;
    for (int l = 0; l < LL; ++l) {
        const __bf16* Wq_l = wq_bf + (size_t)l * DSq;
        const __bf16* Wk_l = wk_bf + (size_t)l * DSq;
        const __bf16* Wv_l = wv_bf + (size_t)l * DSq;
        const __bf16* Wo_l = wo_bf + (size_t)l * DSq;
        const __bf16* W1_l = w1_bf + (size_t)l * 4 * DSq;
        const __bf16* W2_l = w2_bf + (size_t)l * 4 * DSq;

        // ln1 -> xn (bf16)
        ln_kernel<<<MM / 8, dim3(32, 8), 0, stream>>>(x, ln1_w + (size_t)l * DD,
                                                      ln1_b + (size_t)l * DD, xn);
        // q, k row-major bf16; v transposed bf16 [D, M]
        launch_gemm(xn, Wq_l, bq + (size_t)l * DD, nullptr, qbf, MM, DD, DD,
                    ACT_NONE, OUT_BF16, stream);
        launch_gemm(xn, Wk_l, bk + (size_t)l * DD, nullptr, kbf, MM, DD, DD,
                    ACT_NONE, OUT_BF16, stream);
        launch_gemm(xn, Wv_l, bv + (size_t)l * DD, nullptr, vbt, MM, DD, DD,
                    ACT_NONE, OUT_BF16_T, stream);

        // WMMA flash attention -> yb (bf16)
        flash_attn_kernel<<<(BQ * HH * QT_PER_SEQ) / AT_WAVES, dim3(32, AT_WAVES),
                            0, stream>>>(qbf, kbf, vbt, yb);

        // x = x + yb @ Wo^T + bo   (fp32 residual stream)
        launch_gemm(yb, Wo_l, bo + (size_t)l * DD, x, x, MM, DD, DD,
                    ACT_NONE, OUT_F32, stream);

        // ln2 -> xn (bf16)
        ln_kernel<<<MM / 8, dim3(32, 8), 0, stream>>>(x, ln2_w + (size_t)l * DD,
                                                      ln2_b + (size_t)l * DD, xn);
        // h = gelu(xn @ W1^T + b1) (bf16); x = x + h @ W2^T + b2
        launch_gemm(xn, W1_l, b1 + (size_t)l * 4 * DD, nullptr, hb, MM, 4 * DD, DD,
                    ACT_GELU, OUT_BF16, stream);
        launch_gemm(hb, W2_l, b2 + (size_t)l * DD, x, x, MM, DD, 4 * DD,
                    ACT_NONE, OUT_F32, stream);
    }

    // ---- final LN + LM head (fp32 logits) ----
    ln_kernel<<<MM / 8, dim3(32, 8), 0, stream>>>(x, ln_f_w, ln_f_b, xn);
    launch_gemm(xn, wh_bf, nullptr, nullptr, (float*)d_out, MM, VV, DD,
                ACT_NONE, OUT_F32, stream);
}
